// OffsetAttention_47699906789389
// MI455X (gfx1250) — matrix operand order repaired and checked by reading, not compiled, not run on hardware
//
#include <hip/hip_runtime.h>
#include <hip/hip_bf16.h>

// ---------------------------------------------------------------------------
// OffsetAttention for MI455X (gfx1250, wave32, WMMA).
// B=4, N=2048, D=256.  Heavy GEMMs in bf16 WMMA (v_wmma_f32_16x16x32_bf16),
// statistics (softmax, renorm, batchnorm) in f32.
// Renorm folded into V:  O = P @ diag(1/(eps+colsum)) @ V  =  P @ V'.
// Shared WMMA operands staged in LDS via global_load_async_to_lds_b128
// (ASYNCcnt) so each block pulls them from L2 once instead of 8x.
// ---------------------------------------------------------------------------

typedef __bf16 bf16_t;
typedef __attribute__((ext_vector_type(16))) __bf16 v16bf;
typedef __attribute__((ext_vector_type(8)))  float  v8f;

#define BB 4
#define NN 2048
#define DD 256
#define MTOT (BB * NN)   // 8192

// ---- CDNA5 async copy global->LDS (ASYNCcnt-tracked) -----------------------
// LDS byte offset comes from the vdst VGPR; 16 bytes per lane.
__device__ __forceinline__ void async_load_lds_b128(uint32_t lds_byte_off,
                                                    const void* gptr) {
    asm volatile("global_load_async_to_lds_b128 %0, %1, off"
                 :
                 : "v"(lds_byte_off), "v"(gptr)
                 : "memory");
}
__device__ __forceinline__ void wait_async0() {
    asm volatile("s_wait_asynccnt 0x0" ::: "memory");
}
__device__ __forceinline__ uint32_t lds_off_u32(const void* p) {
    // generic LDS pointer: low 32 bits are the LDS byte offset
    return (uint32_t)(uintptr_t)p;
}

// ---- WMMA fragment loads (NT GEMM: A row-major MxK, BT row-major NxK) ------
// A fragment as used here (contiguous k, same as B): lane = h*16+l holds row (m0+l);
//   VGPR0..7 = K {16h..16h+15}  (pairs per VGPR)
__device__ __forceinline__ v16bf load_frag_a(const bf16_t* __restrict__ base,
                                             int ld, int m0, int kc, int lane) {
    const int l = lane & 15, h = lane >> 4;
    const bf16_t* p = base + (size_t)(m0 + l) * ld + kc + 16 * h;
    union { v16bf v; uint4 q[2]; } u;
    u.q[0] = *reinterpret_cast<const uint4*>(p);
    u.q[1] = *reinterpret_cast<const uint4*>(p + 8);
    return u.v;
}
// Same A layout but sourced from an LDS tile (16 rows x ldpad elems).
__device__ __forceinline__ v16bf load_frag_a_lds(const bf16_t* lA, int ldpad,
                                                 int kc, int lane) {
    const int l = lane & 15, h = lane >> 4;
    const bf16_t* p = lA + l * ldpad + kc + 16 * h;
    union { v16bf v; uint4 q[2]; } u;
    u.q[0] = *reinterpret_cast<const uint4*>(p);
    u.q[1] = *reinterpret_cast<const uint4*>(p + 8);
    return u.v;
}
// B fragment as used here (contiguous k, same as A): lanes 0-15 hold K=kc..kc+15 for column n0+lane,
// lanes 16-31 hold K=kc+16..kc+31.  BT is NxK so this is contiguous.
__device__ __forceinline__ v16bf load_frag_bt(const bf16_t* __restrict__ base,
                                              int ld, int n0, int kc, int lane) {
    const int n = lane & 15, h = lane >> 4;
    const bf16_t* p = base + (size_t)(n0 + n) * ld + kc + 16 * h;
    union { v16bf v; uint4 q[2]; } u;
    u.q[0] = *reinterpret_cast<const uint4*>(p);
    u.q[1] = *reinterpret_cast<const uint4*>(p + 8);
    return u.v;
}
__device__ __forceinline__ v8f wmma_bf16(v16bf a, v16bf b, v8f c) {
    return __builtin_amdgcn_wmma_f32_16x16x32_bf16(false, a, false, b,
                                                   (short)0, c, false, false);
}

// ---- tiny prep kernels ------------------------------------------------------
__global__ __launch_bounds__(256) void zero_f32_kernel(float* p, int n) {
    int i = blockIdx.x * 256 + threadIdx.x;
    if (i < n) p[i] = 0.0f;
}
__global__ __launch_bounds__(256) void f32_to_bf16_kernel(const float* __restrict__ x,
                                                          bf16_t* __restrict__ y, int n) {
    int i = blockIdx.x * 256 + threadIdx.x;
    if (i < n) y[i] = (bf16_t)x[i];
}
// WT[n][d] = W[d][n]  (weights stored (in,out) in the reference)
__global__ __launch_bounds__(256) void transpose_w_kernel(const float* __restrict__ W,
                                                          bf16_t* __restrict__ WT) {
    int i = blockIdx.x * 256 + threadIdx.x;   // 256*256 total
    int d = i >> 8, n = i & 255;
    WT[(size_t)n * DD + d] = (bf16_t)W[(size_t)d * DD + n];
}

// ---- projection: Y(bf16, MxDD) = X(bf16) @ WT^T + bias ---------------------
__global__ __launch_bounds__(256) void proj_kernel(const bf16_t* __restrict__ X,
                                                   const bf16_t* __restrict__ WT,
                                                   const float* __restrict__ bias,
                                                   bf16_t* __restrict__ Y) {
    const int lane = threadIdx.x & 31, wave = threadIdx.x >> 5;
    const int m0 = blockIdx.x * 128 + wave * 16;
    const int n0 = blockIdx.y * 64;
    v8f acc[4] = {};
    for (int kc = 0; kc < DD; kc += 32) {
        v16bf a = load_frag_a(X, DD, m0, kc, lane);
#pragma unroll
        for (int j = 0; j < 4; ++j) {
            v16bf b = load_frag_bt(WT, DD, n0 + 16 * j, kc, lane);
            acc[j] = wmma_bf16(a, b, acc[j]);
        }
    }
    const int h = lane >> 4, nl = lane & 15;
#pragma unroll
    for (int j = 0; j < 4; ++j) {
        const int n = n0 + 16 * j + nl;
        const float bv = bias[n];
#pragma unroll
        for (int r = 0; r < 8; ++r) {
            const int m = m0 + 8 * h + r;
            Y[(size_t)m * DD + n] = (bf16_t)(acc[j][r] + bv);
        }
    }
}

// ---- scores + row softmax + column-sum accumulation ------------------------
// One block = 16 query rows of one batch.  The 16x256 Q tile (shared by all
// 8 waves) is async-staged into LDS once; waves cover all 2048 key columns
// with WMMA into 128KB of LDS, then f32 softmax, bf16 P store + colsum atomics.
#define QPAD 264   // 256 elems + 8 pad (528B rows, 16B aligned)
__global__ __launch_bounds__(256) void scores_kernel(const bf16_t* __restrict__ Q,
                                                     const bf16_t* __restrict__ K,
                                                     bf16_t* __restrict__ P,
                                                     float* __restrict__ colsum) {
    __shared__ __align__(16) bf16_t lA[16 * QPAD];
    extern __shared__ float smem[];
    float* S       = smem;                  // [16][2048]
    float* red     = smem + 16 * NN;        // [16][16]
    float* rowstat = red + 256;             // [16]

    const int b  = blockIdx.y;
    const int m0 = blockIdx.x * 16;
    const bf16_t* Qb = Q + (size_t)b * NN * DD;
    const bf16_t* Kb = K + (size_t)b * NN * DD;
    const int lane = threadIdx.x & 31, wave = threadIdx.x >> 5;

    // async-stage Q tile: 16 rows x 512B; each thread copies 2 x 16B
    {
        const int crow = threadIdx.x >> 4;        // 0..15
        const int cseg = threadIdx.x & 15;        // 0..15, 16B each
        const bf16_t* g = Qb + (size_t)(m0 + crow) * DD + cseg * 8;
        const uint32_t l0 = lds_off_u32(&lA[crow * QPAD + cseg * 8]);
        async_load_lds_b128(l0, g);
        async_load_lds_b128(l0 + 256, g + 128);   // second 256B half of row
    }
    wait_async0();
    __syncthreads();

    for (int g = 0; g < 4; ++g) {
        const int n0 = wave * 256 + g * 64;
        v8f acc[4] = {};
        for (int kc = 0; kc < DD; kc += 32) {
            v16bf a = load_frag_a_lds(lA, QPAD, kc, lane);
#pragma unroll
            for (int j = 0; j < 4; ++j) {
                v16bf bf = load_frag_bt(Kb, DD, n0 + 16 * j, kc, lane);
                acc[j] = wmma_bf16(a, bf, acc[j]);
            }
        }
        const int h = lane >> 4, nl = lane & 15;
#pragma unroll
        for (int j = 0; j < 4; ++j)
#pragma unroll
            for (int r = 0; r < 8; ++r)
                S[(8 * h + r) * NN + n0 + 16 * j + nl] = acc[j][r];
    }
    __syncthreads();

    const int r  = threadIdx.x >> 4;   // row 0..15
    const int c0 = threadIdx.x & 15;
    float mx = -1e30f;
    for (int c = c0; c < NN; c += 16) mx = fmaxf(mx, S[r * NN + c]);
    red[r * 16 + c0] = mx;
    __syncthreads();
    if (c0 == 0) {
        float m = red[r * 16];
        for (int i = 1; i < 16; ++i) m = fmaxf(m, red[r * 16 + i]);
        rowstat[r] = m;
    }
    __syncthreads();
    const float rowmax = rowstat[r];
    float s = 0.0f;
    for (int c = c0; c < NN; c += 16) {
        float e = __expf(S[r * NN + c] - rowmax);
        S[r * NN + c] = e;
        s += e;
    }
    __syncthreads();
    red[r * 16 + c0] = s;
    __syncthreads();
    if (c0 == 0) {
        float t = 0.0f;
        for (int i = 0; i < 16; ++i) t += red[r * 16 + i];
        rowstat[r] = 1.0f / t;
    }
    __syncthreads();

    bf16_t* Pb = P + (size_t)b * NN * NN;
    for (int c = threadIdx.x; c < NN; c += 256) {
        float cs = 0.0f;
#pragma unroll 4
        for (int rr = 0; rr < 16; ++rr) {
            float pv = S[rr * NN + c] * rowstat[rr];
            Pb[(size_t)(m0 + rr) * NN + c] = (bf16_t)pv;
            cs += pv;
        }
        atomicAdd(&colsum[b * NN + c], cs);
    }
}

// ---- V' transpose:  VtT[b][d][k] = V[b][k][d] / (eps + colsum[b][k]) -------
__global__ __launch_bounds__(256) void scaleT_kernel(const bf16_t* __restrict__ V,
                                                     const float* __restrict__ colsum,
                                                     bf16_t* __restrict__ VtT) {
    size_t i = (size_t)blockIdx.x * 256 + threadIdx.x;   // B*N*D total
    const int b   = (int)(i / (NN * DD));
    const int rem = (int)(i % (NN * DD));
    const int k = rem >> 8, d = rem & 255;
    const float rsc = 1.0f / (1e-9f + colsum[b * NN + k]);
    VtT[((size_t)b * DD + d) * NN + k] = (bf16_t)((float)V[i] * rsc);
}

// ---- O = P @ V' with fused residual:  R = bf16(values - O) -----------------
// The 64x32 B tile per K-chunk is shared by all 8 waves: async-stage it in
// LDS (4KB, one b128 per thread), double-buffer the A fragment in registers.
#define BPAD 40    // 32 elems + 8 pad per row (80B rows, 16B aligned)
__global__ __launch_bounds__(256) void av_kernel(const bf16_t* __restrict__ Pmat,
                                                 const bf16_t* __restrict__ VtT,
                                                 const float* __restrict__ values,
                                                 bf16_t* __restrict__ Rb) {
    __shared__ __align__(16) bf16_t lB[64 * BPAD];
    const int b = blockIdx.z;
    const int lane = threadIdx.x & 31, wave = threadIdx.x >> 5;
    const int m0 = blockIdx.x * 128 + wave * 16;
    const int n0 = blockIdx.y * 64;
    const bf16_t* A  = Pmat + (size_t)b * NN * NN;
    const bf16_t* BT = VtT  + (size_t)b * DD * NN;

    // async-copy mapping: thread t -> row t>>2 (of 64), 16B segment t&3
    const int crow = threadIdx.x >> 2, cseg = threadIdx.x & 3;
    const bf16_t* gB = BT + (size_t)(n0 + crow) * NN + cseg * 8;
    const uint32_t lB0 = lds_off_u32(&lB[crow * BPAD + cseg * 8]);
    // per-lane A row pointer for prefetch
    const bf16_t* Arow = A + (size_t)(m0 + (lane & 15)) * NN;

    v8f acc[4] = {};
    v16bf a_cur = load_frag_a(A, NN, m0, 0, lane);
    const int nl = lane & 15, h = lane >> 4;

    for (int kc = 0; kc < NN; kc += 32) {
        __syncthreads();                       // prior chunk's LDS reads done
        async_load_lds_b128(lB0, gB + kc);     // stage shared B tile
        v16bf a_nxt = a_cur;
        if (kc + 32 < NN) a_nxt = load_frag_a(A, NN, m0, kc + 32, lane);
        if (kc + 64 < NN) __builtin_prefetch((const void*)(Arow + kc + 64), 0, 1);
        wait_async0();
        __syncthreads();                       // tile visible to all waves
#pragma unroll
        for (int j = 0; j < 4; ++j) {
            const bf16_t* lp = &lB[(16 * j + nl) * BPAD + 16 * h];
            union { v16bf v; uint4 q[2]; } u;
            u.q[0] = *reinterpret_cast<const uint4*>(lp);
            u.q[1] = *reinterpret_cast<const uint4*>(lp + 8);
            acc[j] = wmma_bf16(a_cur, u.v, acc[j]);
        }
        a_cur = a_nxt;
    }

    const float* valb = values + (size_t)b * NN * DD;
    bf16_t* Ro = Rb + (size_t)b * NN * DD;
#pragma unroll
    for (int j = 0; j < 4; ++j)
#pragma unroll
        for (int r = 0; r < 8; ++r) {
            const int m = m0 + 8 * h + r, n = n0 + 16 * j + nl;
            Ro[(size_t)m * DD + n] =
                (bf16_t)(valb[(size_t)m * DD + n] - acc[j][r]);
        }
}

// ---- T = R @ Wo + bo, fused BN statistic accumulation ----------------------
__global__ __launch_bounds__(256) void outproj_kernel(const bf16_t* __restrict__ Rb,
                                                      const bf16_t* __restrict__ WoT,
                                                      const float* __restrict__ bo,
                                                      float* __restrict__ T,
                                                      float* __restrict__ bnsum,
                                                      float* __restrict__ bnsumsq) {
    const int lane = threadIdx.x & 31, wave = threadIdx.x >> 5;
    const int m0 = blockIdx.x * 128 + wave * 16;
    const int n0 = blockIdx.y * 64;
    v8f acc[4] = {};
    for (int kc = 0; kc < DD; kc += 32) {
        v16bf a = load_frag_a(Rb, DD, m0, kc, lane);
#pragma unroll
        for (int j = 0; j < 4; ++j) {
            v16bf bf = load_frag_bt(WoT, DD, n0 + 16 * j, kc, lane);
            acc[j] = wmma_bf16(a, bf, acc[j]);
        }
    }
    const int h = lane >> 4, nl = lane & 15;
#pragma unroll
    for (int j = 0; j < 4; ++j) {
        const int n = n0 + 16 * j + nl;
        const float bias = bo[n];
        float s = 0.0f, s2 = 0.0f;
#pragma unroll
        for (int r = 0; r < 8; ++r) {
            const int m = m0 + 8 * h + r;
            const float y = acc[j][r] + bias;
            T[(size_t)m * DD + n] = y;
            s += y; s2 += y * y;
        }
        atomicAdd(&bnsum[n],   s);
        atomicAdd(&bnsumsq[n], s2);
    }
}

// ---- BatchNorm (training stats) + ReLU finalize ----------------------------
__global__ __launch_bounds__(256) void bn_relu_kernel(const float* __restrict__ T,
                                                      const float* __restrict__ bnsum,
                                                      const float* __restrict__ bnsumsq,
                                                      const float* __restrict__ gamma,
                                                      const float* __restrict__ beta,
                                                      float* __restrict__ out) {
    size_t i = (size_t)blockIdx.x * 256 + threadIdx.x;
    const int n = (int)(i & 255);
    const float inv_cnt = 1.0f / (float)MTOT;
    const float mean = bnsum[n] * inv_cnt;
    const float var  = bnsumsq[n] * inv_cnt - mean * mean;
    const float y = (T[i] - mean) * rsqrtf(var + 1e-5f) * gamma[n] + beta[n];
    out[i] = fmaxf(y, 0.0f);
}

// ---------------------------------------------------------------------------
extern "C" void kernel_launch(void* const* d_in, const int* in_sizes, int n_in,
                              void* d_out, int out_size, void* d_ws, size_t ws_size,
                              hipStream_t stream) {
    const float* queries = (const float*)d_in[0];
    const float* keys    = (const float*)d_in[1];
    const float* values  = (const float*)d_in[2];
    const float* Wq = (const float*)d_in[3];  const float* bq = (const float*)d_in[4];
    const float* Wk = (const float*)d_in[5];  const float* bk = (const float*)d_in[6];
    const float* Wv = (const float*)d_in[7];  const float* bv = (const float*)d_in[8];
    const float* Wo = (const float*)d_in[9];  const float* bo = (const float*)d_in[10];
    const float* gamma = (const float*)d_in[11];
    const float* beta  = (const float*)d_in[12];
    float* out = (float*)d_out;

    char* ws = (char*)d_ws;
    size_t off = 0;
    auto alloc = [&](size_t bytes) -> void* {
        void* p = ws + off;
        off = (off + bytes + 255) & ~(size_t)255;
        return p;
    };
    const size_t ND = (size_t)MTOT * DD;            // 2M elems
    bf16_t* Xq  = (bf16_t*)alloc(ND * 2);
    bf16_t* Xk  = (bf16_t*)alloc(ND * 2);
    bf16_t* Xv  = (bf16_t*)alloc(ND * 2);
    bf16_t* WqT = (bf16_t*)alloc(DD * DD * 2);
    bf16_t* WkT = (bf16_t*)alloc(DD * DD * 2);
    bf16_t* WvT = (bf16_t*)alloc(DD * DD * 2);
    bf16_t* WoT = (bf16_t*)alloc(DD * DD * 2);
    bf16_t* Qb  = (bf16_t*)alloc(ND * 2);
    bf16_t* Kb  = (bf16_t*)alloc(ND * 2);
    bf16_t* Vb  = (bf16_t*)alloc(ND * 2);
    bf16_t* P   = (bf16_t*)alloc((size_t)BB * NN * NN * 2);   // 33.5 MB
    bf16_t* VtT = (bf16_t*)alloc(ND * 2);
    bf16_t* Rb  = (bf16_t*)alloc(ND * 2);
    float*  T   = (float*)alloc(ND * 4);
    float*  stats = (float*)alloc((BB * NN + 2 * DD) * 4);    // colsum|bnsum|bnsumsq
    float* colsum  = stats;
    float* bnsum   = stats + BB * NN;
    float* bnsumsq = stats + BB * NN + DD;
    (void)ws_size; (void)n_in; (void)in_sizes; (void)out_size;

    const int stats_n = BB * NN + 2 * DD;
    zero_f32_kernel<<<(stats_n + 255) / 256, 256, 0, stream>>>(stats, stats_n);

    f32_to_bf16_kernel<<<ND / 256, 256, 0, stream>>>(queries, Xq, (int)ND);
    f32_to_bf16_kernel<<<ND / 256, 256, 0, stream>>>(keys,    Xk, (int)ND);
    f32_to_bf16_kernel<<<ND / 256, 256, 0, stream>>>(values,  Xv, (int)ND);
    transpose_w_kernel<<<DD * DD / 256, 256, 0, stream>>>(Wq, WqT);
    transpose_w_kernel<<<DD * DD / 256, 256, 0, stream>>>(Wk, WkT);
    transpose_w_kernel<<<DD * DD / 256, 256, 0, stream>>>(Wv, WvT);
    transpose_w_kernel<<<DD * DD / 256, 256, 0, stream>>>(Wo, WoT);

    dim3 gproj(MTOT / 128, DD / 64);
    proj_kernel<<<gproj, 256, 0, stream>>>(Xq, WqT, bq, Qb);
    proj_kernel<<<gproj, 256, 0, stream>>>(Xk, WkT, bk, Kb);
    proj_kernel<<<gproj, 256, 0, stream>>>(Xv, WvT, bv, Vb);

    const size_t smem_bytes = (16 * NN + 256 + 16) * sizeof(float);   // ~132 KB LDS
    scores_kernel<<<dim3(NN / 16, BB), 256, smem_bytes, stream>>>(Qb, Kb, P, colsum);

    scaleT_kernel<<<ND / 256, 256, 0, stream>>>(Vb, colsum, VtT);

    av_kernel<<<dim3(NN / 128, DD / 64, BB), 256, 0, stream>>>(P, VtT, values, Rb);

    outproj_kernel<<<gproj, 256, 0, stream>>>(Rb, WoT, bo, T, bnsum, bnsumsq);

    bn_relu_kernel<<<ND / 256, 256, 0, stream>>>(T, bnsum, bnsumsq, gamma, beta, out);
}
